// AudioTransformer_68418829025621
// MI455X (gfx1250) — compile-verified, matrix operand order checked by reading, not run on hardware
//
#include <hip/hip_runtime.h>
#include <hip/hip_bf16.h>
#include <math.h>
#include <stdint.h>

typedef _Float16 f16;
typedef __attribute__((ext_vector_type(8)))  _Float16 v8h;
typedef __attribute__((ext_vector_type(16))) _Float16 v16h;
typedef __attribute__((ext_vector_type(8)))  float    v8f;
typedef __attribute__((ext_vector_type(4)))  int      v4i;

#define DEV __device__ __forceinline__

constexpr int BB = 8, SS = 1024, DD = 512, HH = 8, DHD = 64, LLAYERS = 8, FFD = 2048, NMEL = 80, KPAD = 96;

#if defined(__has_builtin)
# if __has_builtin(__builtin_amdgcn_global_load_async_to_lds_b128)
#  define HAS_ASYNC_LDS 1
# endif
# if __has_builtin(__builtin_amdgcn_update_dpp)
#  define HAS_DPP 1
# endif
#endif
#ifndef HAS_ASYNC_LDS
# define HAS_ASYNC_LDS 0
#endif
#ifndef HAS_DPP
# define HAS_DPP 0
#endif

DEV v8h ld8(const f16* p) { return *(const v8h*)p; }
DEV v16h mk16(v8h a, v8h b) {
  return __builtin_shufflevector(a, b, 0,1,2,3,4,5,6,7,8,9,10,11,12,13,14,15);
}
DEV v8f zero8f() {
  v8f z;
  for (int i = 0; i < 8; i++) z[i] = 0.f;
  return z;
}
DEV v8h zero8h() {
  v8h z;
  for (int i = 0; i < 8; i++) z[i] = (_Float16)0;
  return z;
}
DEV v8f wmma_f16(v16h a, v16h b, v8f c) {
  return __builtin_amdgcn_wmma_f32_16x16x32_f16(false, a, false, b, (short)0, c, false, false);
}

#if HAS_ASYNC_LDS
typedef __attribute__((address_space(1))) v4i g_v4i;
typedef __attribute__((address_space(3))) v4i l_v4i;
DEV void async_cp_b128(const f16* g, const f16* l) {
  __builtin_amdgcn_global_load_async_to_lds_b128(
      (g_v4i*)(uintptr_t)g, (l_v4i*)(uint32_t)(uintptr_t)l, 0, 0);
}
DEV void wait_async0() {
#if __has_builtin(__builtin_amdgcn_s_wait_asynccnt)
  __builtin_amdgcn_s_wait_asynccnt(0);
  asm volatile("" ::: "memory");
#else
  asm volatile("s_wait_asynccnt 0" ::: "memory");
#endif
}
#endif

// Butterfly reductions over the 16-lane half-wave using DPP16.
// xor1 = quad_perm(1,0,3,2)=0xB1, xor2 = quad_perm(2,3,0,1)=0x4E,
// xor7 = row_half_mirror (0x141), xor15 = row_mirror (0x140).
// {1,2,7,15} are linearly independent -> full 16-lane reduction.
#if HAS_DPP
template<int CTRL>
DEV float dpp_mov(float v) {
  return __int_as_float(__builtin_amdgcn_update_dpp(0, __float_as_int(v), CTRL, 0xF, 0xF, true));
}
DEV float red_max16(float v) {
  v = fmaxf(v, dpp_mov<0xB1>(v));
  v = fmaxf(v, dpp_mov<0x4E>(v));
  v = fmaxf(v, dpp_mov<0x141>(v));
  v = fmaxf(v, dpp_mov<0x140>(v));
  return v;
}
DEV float red_sum16(float v) {
  v += dpp_mov<0xB1>(v);
  v += dpp_mov<0x4E>(v);
  v += dpp_mov<0x141>(v);
  v += dpp_mov<0x140>(v);
  return v;
}
#else
DEV float red_max16(float v) {
  for (int o = 8; o >= 1; o >>= 1) v = fmaxf(v, __shfl_xor(v, o, 16));
  return v;
}
DEV float red_sum16(float v) {
  for (int o = 8; o >= 1; o >>= 1) v += __shfl_xor(v, o, 16);
  return v;
}
#endif

// ---------------------------------------------------------------------------
// Generic WMMA GEMM:  C(MxN) = A(MxK,f16) * B(KxN)  with B given transposed
// as BT(NxK,f16).  EPI bitmask: 1=+bias, 2=+residual(f32, ld=N, row%resMod),
// 4=exact GELU, 8=store f16 (else f32).
// Block tile 128x128, 256 threads = 8 waves, each wave 32x64 (2x4 frags).
// K-tiles are double-buffered through LDS via async global->LDS copies
// (ASYNCcnt) when available; sync fallback prefetches the next tile.
// ---------------------------------------------------------------------------
template<int EPI>
__global__ __launch_bounds__(256) void gemm_wmma_kernel(
    const f16* __restrict__ A, const f16* __restrict__ BT,
    const float* __restrict__ bias, const float* __restrict__ res,
    float* __restrict__ oF32, f16* __restrict__ oF16,
    int M, int N, int K, int resMod)
{
  __shared__ __align__(16) f16 sA[2][128 * 32];
  __shared__ __align__(16) f16 sB[2][128 * 32];
  const int tid  = threadIdx.x;
  const int lane = tid & 31;
  const int wave = tid >> 5;
  const int l15  = lane & 15;
  const int off  = (lane < 16) ? 0 : 8;
  const int wrow = (wave & 3) * 32;
  const int wcol = (wave >> 2) * 64;
  const int rowBase = blockIdx.y * 128;
  const int colBase = blockIdx.x * 128;

  v8f acc[2][4];
  #pragma unroll
  for (int i = 0; i < 2; i++)
    #pragma unroll
    for (int j = 0; j < 4; j++) acc[i][j] = zero8f();

  auto compute = [&](int buf) {
    v16h av[2], bv[4];
    #pragma unroll
    for (int i = 0; i < 2; i++) {
      const f16* p = &sA[buf][(wrow + i * 16 + l15) * 32 + off];
      av[i] = mk16(ld8(p), ld8(p + 16));
    }
    #pragma unroll
    for (int j = 0; j < 4; j++) {
      const f16* p = &sB[buf][(wcol + j * 16 + l15) * 32 + off];
      bv[j] = mk16(ld8(p), ld8(p + 16));
    }
    #pragma unroll
    for (int i = 0; i < 2; i++)
      #pragma unroll
      for (int j = 0; j < 4; j++)
        acc[i][j] = wmma_f16(av[i], bv[j], acc[i][j]);
  };

#if HAS_ASYNC_LDS
  if (colBase + 128 <= N) {
    // ---- async double-buffered pipeline ----
    for (int c = tid; c < 512; c += 256) {
      int r = c >> 2, kc = (c & 3) << 3;
      async_cp_b128(A  + (size_t)(rowBase + r) * K + kc, &sA[0][r * 32 + kc]);
      async_cp_b128(BT + (size_t)(colBase + r) * K + kc, &sB[0][r * 32 + kc]);
    }
    int cur = 0;
    for (int k0 = 0; k0 < K; k0 += 32, cur ^= 1) {
      wait_async0();          // this wave's async writes (incl. buf cur) done
      __syncthreads();        // everyone's writes visible; prior reads retired
      if (k0 + 32 < K) {
        for (int c = tid; c < 512; c += 256) {
          int r = c >> 2, kc = (c & 3) << 3;
          async_cp_b128(A  + (size_t)(rowBase + r) * K + k0 + 32 + kc, &sA[cur ^ 1][r * 32 + kc]);
          async_cp_b128(BT + (size_t)(colBase + r) * K + k0 + 32 + kc, &sB[cur ^ 1][r * 32 + kc]);
        }
      }
      compute(cur);
    }
  } else
#endif
  {
    for (int k0 = 0; k0 < K; k0 += 32) {
      for (int c = tid; c < 512; c += 256) {
        int r = c >> 2, kc = (c & 3) << 3;
        *(v8h*)&sA[0][r * 32 + kc] = ld8(A + (size_t)(rowBase + r) * K + k0 + kc);
        v8h bb = zero8h();
        if (colBase + r < N) bb = ld8(BT + (size_t)(colBase + r) * K + k0 + kc);
        *(v8h*)&sB[0][r * 32 + kc] = bb;
        if (k0 + 32 < K) {
          __builtin_prefetch(A + (size_t)(rowBase + r) * K + k0 + 32 + kc, 0, 3);
          if (colBase + r < N)
            __builtin_prefetch(BT + (size_t)(colBase + r) * K + k0 + 32 + kc, 0, 3);
        }
      }
      __syncthreads();
      compute(0);
      __syncthreads();
    }
  }

  #pragma unroll
  for (int i = 0; i < 2; i++)
    #pragma unroll
    for (int j = 0; j < 4; j++)
      #pragma unroll
      for (int r = 0; r < 8; r++) {
        int row = rowBase + wrow + i * 16 + ((lane < 16) ? r : r + 8);
        int col = colBase + wcol + j * 16 + l15;
        if (col < N) {
          float v = acc[i][j][r];
          if (EPI & 1) v += bias[col];
          if (EPI & 2) { int rr = resMod ? (row % resMod) : row; v += res[(size_t)rr * N + col]; }
          if (EPI & 4) v = 0.5f * v * (1.0f + erff(v * 0.70710678118f));
          if (EPI & 8) oF16[(size_t)row * N + col] = (f16)v;
          else         oF32[(size_t)row * N + col] = v;
        }
      }
}

// ---------------------------------------------------------------------------
// Flash attention: one wave per 16-query tile per (b,h). Q/K in (B,S,D) f16
// (head h at column h*64); V pre-transposed to (B,H,DH,S) f16. Online softmax
// with DPP16 butterfly reductions; scores and P*V through WMMA, 32 keys/iter.
// ---------------------------------------------------------------------------
__global__ __launch_bounds__(128) void flash_attn_kernel(
    const f16* __restrict__ q, const f16* __restrict__ k,
    const f16* __restrict__ vt, f16* __restrict__ ctx)
{
  __shared__ __align__(16) f16 sP[4][16 * 32];
  const int tid  = threadIdx.x;
  const int w    = tid >> 5;
  const int lane = tid & 31;
  const int wid  = blockIdx.x * 4 + w;
  const int qt   = wid & 63;          // S/16 = 64 q-tiles
  const int h    = (wid >> 6) & 7;
  const int b    = wid >> 9;
  const int l15  = lane & 15;
  const int off  = (lane < 16) ? 0 : 8;

  const f16* qp = q + ((size_t)(b * SS + qt * 16 + l15) * DD + h * DHD);
  v16h qa0 = mk16(ld8(qp + off),      ld8(qp + off + 16));
  v16h qa1 = mk16(ld8(qp + 32 + off), ld8(qp + 48 + off));

  const f16* kb = k + (size_t)b * SS * DD + h * DHD;
  const f16* vb = vt + ((size_t)(b * HH + h) * DHD) * SS;

  float m[8], lsum[8], alpha[8];
  v8f c[4];
  #pragma unroll
  for (int r = 0; r < 8; r++) { m[r] = -3.0e38f; lsum[r] = 0.f; }
  #pragma unroll
  for (int j = 0; j < 4; j++) c[j] = zero8f();

  for (int kt = 0; kt < SS; kt += 32) {
    const f16* p00 = kb + (size_t)(kt + l15) * DD + off;
    const f16* p10 = kb + (size_t)(kt + 16 + l15) * DD + off;
    v8f s0 = zero8f(), s1 = zero8f();
    s0 = wmma_f16(qa0, mk16(ld8(p00),      ld8(p00 + 16)), s0);
    s0 = wmma_f16(qa1, mk16(ld8(p00 + 32), ld8(p00 + 48)), s0);
    s1 = wmma_f16(qa0, mk16(ld8(p10),      ld8(p10 + 16)), s1);
    s1 = wmma_f16(qa1, mk16(ld8(p10 + 32), ld8(p10 + 48)), s1);

    #pragma unroll
    for (int r = 0; r < 8; r++) {
      float a0 = s0[r] * 0.125f, a1 = s1[r] * 0.125f;   // 1/sqrt(64)
      float mx = red_max16(fmaxf(a0, a1));
      float mn = fmaxf(m[r], mx);
      float al = __expf(m[r] - mn);
      float p0 = __expf(a0 - mn), p1 = __expf(a1 - mn);
      float sm = red_sum16(p0 + p1);
      lsum[r] = lsum[r] * al + sm;
      m[r] = mn; alpha[r] = al;
      int row = (lane < 16) ? r : r + 8;
      sP[w][row * 32 + l15]      = (f16)p0;
      sP[w][row * 32 + 16 + l15] = (f16)p1;
    }
    #pragma unroll
    for (int j = 0; j < 4; j++)
      #pragma unroll
      for (int r = 0; r < 8; r++) c[j][r] *= alpha[r];

    asm volatile("s_wait_dscnt 0" ::: "memory");   // P stores visible to reload
    const f16* pp = &sP[w][l15 * 32 + off];
    v16h pa = mk16(ld8(pp), ld8(pp + 16));
    #pragma unroll
    for (int j = 0; j < 4; j++) {
      const f16* vp = vb + (size_t)(j * 16 + l15) * SS + kt + off;
      c[j] = wmma_f16(pa, mk16(ld8(vp), ld8(vp + 16)), c[j]);
    }
  }

  f16* cp = ctx + ((size_t)(b * SS + qt * 16) * DD + h * DHD);
  #pragma unroll
  for (int r = 0; r < 8; r++) {
    float inv = 1.0f / lsum[r];
    int row = (lane < 16) ? r : r + 8;
    #pragma unroll
    for (int j = 0; j < 4; j++)
      cp[(size_t)row * DD + j * 16 + l15] = (f16)(c[j][r] * inv);
  }
}

// ---------------------------------------------------------------------------
// Support kernels
// ---------------------------------------------------------------------------
__global__ void layernorm_kernel(const float* __restrict__ in, const float* __restrict__ g,
                                 const float* __restrict__ bt, f16* __restrict__ out)
{
  __shared__ float s1[8], s2[8];
  int row = blockIdx.x;
  const float* x = in + (size_t)row * DD;
  float a = x[threadIdx.x], b = x[threadIdx.x + 256];
  float sum = a + b, sq = a * a + b * b;
  #pragma unroll
  for (int o = 16; o >= 1; o >>= 1) { sum += __shfl_xor(sum, o, 32); sq += __shfl_xor(sq, o, 32); }
  if ((threadIdx.x & 31) == 0) { s1[threadIdx.x >> 5] = sum; s2[threadIdx.x >> 5] = sq; }
  __syncthreads();
  float ts = 0.f, tq = 0.f;
  #pragma unroll
  for (int i = 0; i < 8; i++) { ts += s1[i]; tq += s2[i]; }
  float mu = ts / DD;
  float rs = rsqrtf(tq / DD - mu * mu + 1e-5f);
  out[(size_t)row * DD + threadIdx.x]       = (f16)((a - mu) * rs * g[threadIdx.x] + bt[threadIdx.x]);
  out[(size_t)row * DD + threadIdx.x + 256] = (f16)((b - mu) * rs * g[threadIdx.x + 256] + bt[threadIdx.x + 256]);
}

__global__ void wT_kernel(const float* __restrict__ in, f16* __restrict__ out,
                          int Kk, int Nn, long total)  // (L,K,N)f32 -> (L,N,K)f16
{
  long i = (long)blockIdx.x * 256 + threadIdx.x;
  if (i >= total) return;
  long kn = (long)Kk * Nn;
  long bb = i / kn, r = i % kn;
  int kk = (int)(r / Nn), nn = (int)(r % Nn);
  out[bb * kn + (long)nn * Kk + kk] = (f16)in[i];
}

__global__ void mel_pad_kernel(const float* __restrict__ in, f16* __restrict__ out)
{
  int i = blockIdx.x * 256 + threadIdx.x;
  if (i >= BB * SS * KPAD) return;
  int r = i / KPAD, kk = i % KPAD;
  out[i] = (kk < NMEL) ? (f16)in[r * NMEL + kk] : (f16)0.f;
}

__global__ void projwT_kernel(const float* __restrict__ in, f16* __restrict__ out) // (80,512)->(512,96)
{
  int i = blockIdx.x * 256 + threadIdx.x;
  if (i >= DD * KPAD) return;
  int n = i / KPAD, kk = i % KPAD;
  out[i] = (kk < NMEL) ? (f16)in[kk * DD + n] : (f16)0.f;
}

__global__ void vT_kernel(const f16* __restrict__ v, f16* __restrict__ vt) // (B,S,D)->(B,H,DH,S)
{
  int i = blockIdx.x * 256 + threadIdx.x;
  if (i >= BB * SS * DD) return;
  int b = i / (SS * DD), r = i % (SS * DD);
  int s = r / DD, cc = r % DD;
  int h = cc / DHD, d = cc % DHD;
  vt[((size_t)(b * HH + h) * DHD + d) * SS + s] = v[i];
}

__global__ void pool_kernel(const f16* __restrict__ hN, float* __restrict__ pooled)
{
  int i = blockIdx.x * 256 + threadIdx.x;
  if (i >= BB * DD) return;
  int b = i / DD, d = i % DD;
  float s = 0.f;
  for (int t = 0; t < SS; t++) s += (float)hN[((size_t)b * SS + t) * DD + d];
  pooled[i] = s * (1.0f / SS);
}

__global__ void small_fc_kernel(const float* __restrict__ x, const float* __restrict__ w,
                                const float* __restrict__ bias, float* __restrict__ out, int N)
{
  int i = blockIdx.x * 256 + threadIdx.x;
  if (i >= BB * N) return;
  int b = i / N, n = i % N;
  float s = bias[n];
  for (int kk = 0; kk < DD; kk++) s += x[b * DD + kk] * w[(size_t)kk * N + n];
  out[i] = s;
}

// ---------------------------------------------------------------------------
extern "C" void kernel_launch(void* const* d_in, const int* in_sizes, int n_in,
                              void* d_out, int out_size, void* d_ws, size_t ws_size,
                              hipStream_t stream)
{
  (void)in_sizes; (void)n_in; (void)out_size; (void)ws_size;
  const float* mel   = (const float*)d_in[0];
  const float* pe    = (const float*)d_in[1];
  const float* projw = (const float*)d_in[2];
  const float* projb = (const float*)d_in[3];
  const float* Wq = (const float*)d_in[4];  const float* bq = (const float*)d_in[5];
  const float* Wk = (const float*)d_in[6];  const float* bk = (const float*)d_in[7];
  const float* Wv = (const float*)d_in[8];  const float* bv = (const float*)d_in[9];
  const float* Wo = (const float*)d_in[10]; const float* bo = (const float*)d_in[11];
  const float* ln1g = (const float*)d_in[12]; const float* ln1b = (const float*)d_in[13];
  const float* ln2g = (const float*)d_in[14]; const float* ln2b = (const float*)d_in[15];
  const float* Wi = (const float*)d_in[16]; const float* bi = (const float*)d_in[17];
  const float* Wf = (const float*)d_in[18]; const float* bf = (const float*)d_in[19];
  const float* lnfg = (const float*)d_in[20]; const float* lnfb = (const float*)d_in[21];
  const float* langw = (const float*)d_in[22]; const float* langb = (const float*)d_in[23];
  const float* intw  = (const float*)d_in[24]; const float* intb  = (const float*)d_in[25];
  const float* emow  = (const float*)d_in[26]; const float* emob  = (const float*)d_in[27];
  const float* spw   = (const float*)d_in[28]; const float* spb   = (const float*)d_in[29];

  const int M = BB * SS; // 8192
  char* p = (char*)d_ws;
  auto alloc = [&](size_t bytes) { char* r = p; p += (bytes + 255) & ~(size_t)255; return r; };
  f16*   melP = (f16*)alloc((size_t)M * KPAD * 2);
  f16*   pWT  = (f16*)alloc((size_t)DD * KPAD * 2);
  f16*   WqT  = (f16*)alloc((size_t)LLAYERS * DD * DD * 2);
  f16*   WkT  = (f16*)alloc((size_t)LLAYERS * DD * DD * 2);
  f16*   WvT  = (f16*)alloc((size_t)LLAYERS * DD * DD * 2);
  f16*   WoT  = (f16*)alloc((size_t)LLAYERS * DD * DD * 2);
  f16*   WiT  = (f16*)alloc((size_t)LLAYERS * DD * FFD * 2);
  f16*   WfT  = (f16*)alloc((size_t)LLAYERS * FFD * DD * 2);
  f16*   spWT = (f16*)alloc((size_t)1000 * DD * 2);
  float* hbuf = (float*)alloc((size_t)M * DD * 4);
  f16*   xb   = (f16*)alloc((size_t)M * DD * 2);
  f16*   qb   = (f16*)alloc((size_t)M * DD * 2);
  f16*   kbf  = (f16*)alloc((size_t)M * DD * 2);
  f16*   vbf  = (f16*)alloc((size_t)M * DD * 2);
  f16*   vtb  = (f16*)alloc((size_t)M * DD * 2);
  f16*   ctxb = (f16*)alloc((size_t)M * DD * 2);
  f16*   ffb  = (f16*)alloc((size_t)M * FFD * 2);
  f16*   hN   = (f16*)alloc((size_t)M * DD * 2);
  float* pooled = (float*)alloc((size_t)BB * DD * 4);

  dim3 blk(256);
  // ---- weight / input conversion (f32 -> f16, transposed to NxK) ----
  mel_pad_kernel<<<(M * KPAD + 255) / 256, blk, 0, stream>>>(mel, melP);
  projwT_kernel<<<(DD * KPAD + 255) / 256, blk, 0, stream>>>(projw, pWT);
  long tw = (long)LLAYERS * DD * DD;
  wT_kernel<<<(tw + 255) / 256, blk, 0, stream>>>(Wq, WqT, DD, DD, tw);
  wT_kernel<<<(tw + 255) / 256, blk, 0, stream>>>(Wk, WkT, DD, DD, tw);
  wT_kernel<<<(tw + 255) / 256, blk, 0, stream>>>(Wv, WvT, DD, DD, tw);
  wT_kernel<<<(tw + 255) / 256, blk, 0, stream>>>(Wo, WoT, DD, DD, tw);
  long ti = (long)LLAYERS * DD * FFD;
  wT_kernel<<<(ti + 255) / 256, blk, 0, stream>>>(Wi, WiT, DD, FFD, ti);
  wT_kernel<<<(ti + 255) / 256, blk, 0, stream>>>(Wf, WfT, FFD, DD, ti);
  long tsw = (long)DD * 1000;
  wT_kernel<<<(tsw + 255) / 256, blk, 0, stream>>>(spw, spWT, DD, 1000, tsw);

  // ---- input projection + positional encoding (residual = pe[row % S]) ----
  gemm_wmma_kernel<3><<<dim3(DD / 128, M / 128), blk, 0, stream>>>(
      melP, pWT, projb, pe, hbuf, nullptr, M, DD, KPAD, SS);

  // ---- transformer layers ----
  for (int l = 0; l < LLAYERS; l++) {
    layernorm_kernel<<<M, blk, 0, stream>>>(hbuf, ln1g + l * DD, ln1b + l * DD, xb);
    gemm_wmma_kernel<9><<<dim3(DD / 128, M / 128), blk, 0, stream>>>(
        xb, WqT + (size_t)l * DD * DD, bq + l * DD, nullptr, nullptr, qb, M, DD, DD, 0);
    gemm_wmma_kernel<9><<<dim3(DD / 128, M / 128), blk, 0, stream>>>(
        xb, WkT + (size_t)l * DD * DD, bk + l * DD, nullptr, nullptr, kbf, M, DD, DD, 0);
    gemm_wmma_kernel<9><<<dim3(DD / 128, M / 128), blk, 0, stream>>>(
        xb, WvT + (size_t)l * DD * DD, bv + l * DD, nullptr, nullptr, vbf, M, DD, DD, 0);
    vT_kernel<<<(M * DD + 255) / 256, blk, 0, stream>>>(vbf, vtb);
    flash_attn_kernel<<<(BB * HH * (SS / 16)) / 4, 128, 0, stream>>>(qb, kbf, vtb, ctxb);
    gemm_wmma_kernel<3><<<dim3(DD / 128, M / 128), blk, 0, stream>>>(
        ctxb, WoT + (size_t)l * DD * DD, bo + l * DD, hbuf, hbuf, nullptr, M, DD, DD, 0);
    layernorm_kernel<<<M, blk, 0, stream>>>(hbuf, ln2g + l * DD, ln2b + l * DD, xb);
    gemm_wmma_kernel<13><<<dim3(FFD / 128, M / 128), blk, 0, stream>>>(
        xb, WiT + (size_t)l * DD * FFD, bi + l * FFD, nullptr, nullptr, ffb, M, FFD, DD, 0);
    gemm_wmma_kernel<3><<<dim3(DD / 128, M / 128), blk, 0, stream>>>(
        ffb, WfT + (size_t)l * FFD * DD, bf + l * DD, hbuf, hbuf, nullptr, M, DD, FFD, 0);
  }

  // ---- final LN, pooling, heads ----
  layernorm_kernel<<<M, blk, 0, stream>>>(hbuf, lnfg, lnfb, hN);
  pool_kernel<<<(BB * DD + 255) / 256, blk, 0, stream>>>(hN, pooled);
  float* out = (float*)d_out;
  small_fc_kernel<<<(BB * 100 + 255) / 256, blk, 0, stream>>>(pooled, langw, langb, out, 100);
  small_fc_kernel<<<(BB * 50 + 255) / 256, blk, 0, stream>>>(pooled, intw, intb, out + 800, 50);
  small_fc_kernel<<<(BB * 8 + 255) / 256, blk, 0, stream>>>(pooled, emow, emob, out + 1200, 8);
  gemm_wmma_kernel<1><<<dim3((1000 + 127) / 128, M / 128), blk, 0, stream>>>(
      hN, spWT, spb, nullptr, out + 1264, nullptr, M, 1000, DD, 0);
}
